// Net_25890062861056
// MI455X (gfx1250) — matrix operand order repaired and checked by reading, not compiled, not run on hardware
//
#include <hip/hip_runtime.h>
#include <hip/hip_bf16.h>
#include <math.h>

#define NN 100000   // nodes
#define NE 1600000  // edges
#define NG 500      // graphs
#define FD 128      // F_IN == HID
#define NC 10       // classes

typedef __bf16 bf16_t;
typedef __attribute__((ext_vector_type(16))) __bf16 v16bf;
typedef __attribute__((ext_vector_type(4)))  __bf16 v4bf;
typedef __attribute__((ext_vector_type(8)))  float  v8f;

// ---------------------------------------------------------------------------
// Pack B (f32, KxN row-major) into WMMA fragment-major bf16 layout:
// for tile t=n/16, kstep ks=k/32, lane l: the 16 bf16 values lane l feeds into
// v_wmma (col = t*16 + (l&15), K = ks*32 + (l>>4)*16 + e) are contiguous.
// Each GEMM lane then loads its whole B fragment as 32 contiguous bytes.
// ---------------------------------------------------------------------------
__global__ void pack_b_bf16(const float* __restrict__ W, bf16_t* __restrict__ out,
                            int K, int N, int kSteps /* = K/32 */) {
    int id = blockIdx.x * blockDim.x + threadIdx.x;
    if (id >= K * N) return;
    int e    = id & 15;
    int lane = (id >> 4) & 31;
    int rest = id >> 9;
    int ks   = rest % kSteps;
    int tile = rest / kSteps;
    int col  = tile * 16 + (lane & 15);
    int kk   = ks * 32 + (lane >> 4) * 16 + e;
    out[id] = (bf16_t)W[(size_t)kk * N + col];
}

// ---------------------------------------------------------------------------
// WMMA GEMM: Y[M,N] = A[M,K](f32) * Bp(packed bf16) (+bias, optional relu)
// block = 256 threads = 8 waves; block tile = 16 rows x 128 cols (wave = 16x16)
// Templated on K so the WMMA loop fully unrolls.  N % 16 == 0.
// ---------------------------------------------------------------------------
template <int K, bool BIAS, bool RELU>
__global__ __launch_bounds__(256)
void wmma_gemm(const float* __restrict__ A, const v16bf* __restrict__ Bp,
               float* __restrict__ Y, int M, int N, const float* __restrict__ bias) {
    constexpr int KSTEPS = K / 32;
    __shared__ bf16_t Xs[16][K + 8];   // padded to keep 8B alignment + dodge conflicts
    const int tid  = threadIdx.x;
    const int row0 = blockIdx.x * 16;

    // stage A tile: float4 global loads -> bf16x4 LDS stores
    constexpr int TOT4 = 16 * K / 4;
#pragma unroll
    for (int idx = tid; idx < TOT4; idx += 256) {
        int r  = idx / (K / 4);
        int c4 = (idx - r * (K / 4)) * 4;
        int gr = row0 + r;
        float4 v = make_float4(0.f, 0.f, 0.f, 0.f);
        if (gr < M) v = *(const float4*)(A + (size_t)gr * K + c4);
        v4bf w; w[0] = (bf16_t)v.x; w[1] = (bf16_t)v.y; w[2] = (bf16_t)v.z; w[3] = (bf16_t)v.w;
        *(v4bf*)&Xs[r][c4] = w;
    }
    __syncthreads();

    const int wave = tid >> 5;
    const int lane = tid & 31;
    const int col0 = blockIdx.y * 128 + wave * 16;
    if (col0 >= N) return;                 // wave-uniform: EXEC stays full

    const int half = lane >> 4;            // 0: lanes 0-15, 1: lanes 16-31
    const int rl   = lane & 15;            // A row (and B col) within tile
    const int tileG = col0 >> 4;

    v8f acc = {0.f, 0.f, 0.f, 0.f, 0.f, 0.f, 0.f, 0.f};
#pragma unroll
    for (int ks = 0; ks < KSTEPS; ++ks) {
        const int kb = ks * 32;
        v16bf a;
        // A 16x32 bf16 layout (matches packed B): lanes<16 -> K0..15; lanes>=16 -> K16..31
#pragma unroll
        for (int e = 0; e < 8; ++e) a[e]     = Xs[rl][kb + half * 16 + e];
#pragma unroll
        for (int e = 0; e < 8; ++e) a[8 + e] = Xs[rl][kb + half * 16 + 8 + e];
        // B fragment: 32 contiguous bytes per lane (pre-packed)
        v16bf b = Bp[(size_t)(tileG * KSTEPS + ks) * 32 + lane];
        acc = __builtin_amdgcn_wmma_f32_16x16x32_bf16(
            false, a, false, b, (short)0, acc, false, false);
    }

    // C/D 16x16 f32: VGPR v -> row v (lanes<16) or v+8 (lanes>=16), col = lane&15
    const int cg = col0 + rl;
    const float bv = BIAS ? bias[cg] : 0.0f;
    if (row0 + 16 <= M) {                  // fast path: whole tile in range
#pragma unroll
        for (int v = 0; v < 8; ++v) {
            float val = acc[v] + bv;
            if (RELU) val = fmaxf(val, 0.0f);
            Y[(size_t)(row0 + v + half * 8) * N + cg] = val;
        }
    } else {
#pragma unroll
        for (int v = 0; v < 8; ++v) {
            int rg = row0 + v + half * 8;
            if (rg < M) {
                float val = acc[v] + bv;
                if (RELU) val = fmaxf(val, 0.0f);
                Y[(size_t)rg * N + cg] = val;
            }
        }
    }
}

// ---------------------------------------------------------------------------
// small utility kernels
// ---------------------------------------------------------------------------
__global__ void fill_f32(float* __restrict__ p, float v, int n) {
    int i = blockIdx.x * blockDim.x + threadIdx.x;
    if (i < n) p[i] = v;
}

// starts[g] = lower_bound(batch, g); counts[g] = lb(g+1) - lb(g)  (batch sorted)
__global__ void graph_ranges(const int* __restrict__ batch, int n,
                             int* __restrict__ starts, int* __restrict__ counts) {
    int g = blockIdx.x * blockDim.x + threadIdx.x;
    if (g >= NG) return;
    int s, e;
    {
        int lo = 0, hi = n;
        while (lo < hi) { int m = (lo + hi) >> 1; if (batch[m] < g) lo = m + 1; else hi = m; }
        s = lo;
    }
    {
        int lo = 0, hi = n;
        while (lo < hi) { int m = (lo + hi) >> 1; if (batch[m] < g + 1) lo = m + 1; else hi = m; }
        e = lo;
    }
    starts[g] = s;
    counts[g] = e - s;
}

// deg[dst] += mask[src]*mask[dst]
__global__ void edge_deg(const int* __restrict__ src, const int* __restrict__ dst,
                         const float* __restrict__ mask, float* __restrict__ deg, int ne) {
    int e = blockIdx.x * blockDim.x + threadIdx.x;
    if (e >= ne) return;
    int s = src[e], d = dst[e];
    float em = mask[s] * mask[d];
    if (em != 0.0f) atomicAdd(&deg[d], em);
}

// deg -> dinv in place: dinv = deg+mask > 0 ? rsqrt(max(deg+mask,1e-12)) : 0
__global__ void node_dinv(float* __restrict__ deg, const float* __restrict__ mask, int n) {
    int i = blockIdx.x * blockDim.x + threadIdx.x;
    if (i >= n) return;
    float dg = deg[i] + mask[i];
    deg[i] = (dg > 0.0f) ? rsqrtf(fmaxf(dg, 1e-12f)) : 0.0f;
}

// agg[dst,:] += h[src,:] * dinv[src]*dinv[dst]*mask[src]*mask[dst]
// one wave per edge; lane 0 computes the coefficient, b128 loads + 4 atomics/lane
__global__ __launch_bounds__(256)
void edge_agg(const int* __restrict__ src, const int* __restrict__ dst,
              const float* __restrict__ dinv, const float* __restrict__ mask,
              const float* __restrict__ h, float* __restrict__ agg, int ne) {
    int lane = threadIdx.x & 31;
    int e = blockIdx.x * 8 + (threadIdx.x >> 5);
    if (e >= ne) return;                   // wave-uniform exit
    int s = 0, d = 0;
    float c = 0.0f;
    if (lane == 0) {
        s = src[e]; d = dst[e];
        c = dinv[s] * dinv[d] * mask[s] * mask[d];
    }
    c = __shfl(c, 0, 32);
    s = __shfl(s, 0, 32);
    d = __shfl(d, 0, 32);
    if (c != 0.0f) {                       // wave-uniform: skip dead edges entirely
        const float4 hv = *(const float4*)(h + (size_t)s * FD + lane * 4);
        float* ap = agg + (size_t)d * FD + lane * 4;
        atomicAdd(ap + 0, hv.x * c);
        atomicAdd(ap + 1, hv.y * c);
        atomicAdd(ap + 2, hv.z * c);
        atomicAdd(ap + 3, hv.w * c);
    }
}

// agg <- relu(agg + h*(dinv^2*mask) + bias)
__global__ void gcn_finalize(float* __restrict__ agg, const float* __restrict__ h,
                             const float* __restrict__ dinv, const float* __restrict__ mask,
                             const float* __restrict__ bias, int n) {
    int id = blockIdx.x * blockDim.x + threadIdx.x;
    if (id >= n) return;
    int i = id >> 7, f = id & 127;
    float dv = dinv[i];
    float v = agg[id] + h[id] * (dv * dv * mask[i]) + bias[f];
    agg[id] = fmaxf(v, 0.0f);
}

__global__ void pnorm_kernel(const float* __restrict__ p, float* __restrict__ out) {
    __shared__ float s[128];
    int t = threadIdx.x;
    float v = p[t];
    s[t] = v * v; __syncthreads();
    for (int o = 64; o > 0; o >>= 1) { if (t < o) s[t] += s[t + o]; __syncthreads(); }
    if (t == 0) out[0] = sqrtf(s[0]);
}

// score[i] = dot(h[i,:], p) / (||p|| + 1e-12); one wave per node
__global__ __launch_bounds__(256)
void score_kernel(const float* __restrict__ h, const float* __restrict__ p,
                  const float* __restrict__ pn, float* __restrict__ score, int n) {
    int wave = threadIdx.x >> 5, lane = threadIdx.x & 31;
    int i = blockIdx.x * 8 + wave;
    if (i >= n) return;
    const float4 hv = *(const float4*)(h + (size_t)i * FD + lane * 4);
    const float4 pv = *(const float4*)(p + lane * 4);
    float d = hv.x * pv.x + hv.y * pv.y + hv.z * pv.z + hv.w * pv.w;
    for (int o = 16; o > 0; o >>= 1) d += __shfl_xor(d, o, 32);
    if (lane == 0) score[i] = d / (pn[0] + 1e-12f);
}

// per-graph top-k selection: keep node if rank-by-score (among kept) < ceil(0.5*kept)
__global__ __launch_bounds__(256)
void pool_kernel(const float* __restrict__ score, const float* __restrict__ mask,
                 const int* __restrict__ starts, const int* __restrict__ counts,
                 float* __restrict__ newmask) {
    __shared__ int red[256];
    __shared__ int kq;
    int g = blockIdx.x;
    int s = starts[g], cnt = counts[g];
    int t = threadIdx.x;
    int kept = 0;
    for (int i = s + t; i < s + cnt; i += 256) kept += (mask[i] > 0.5f);
    red[t] = kept; __syncthreads();
    for (int o = 128; o > 0; o >>= 1) { if (t < o) red[t] += red[t + o]; __syncthreads(); }
    if (t == 0) kq = (int)ceilf(0.5f * (float)red[0]);
    __syncthreads();
    int k = kq;
    for (int i = s + t; i < s + cnt; i += 256) {
        float nm = 0.0f;
        if (mask[i] > 0.5f) {
            float si = score[i];
            int rank = 0;
            for (int j = s; j < s + cnt; ++j) {
                if (mask[j] > 0.5f) {
                    float sj = score[j];
                    rank += (sj > si) || (sj == si && j < i);   // stable tie-break
                }
            }
            nm = (rank < k) ? 1.0f : 0.0f;
        }
        newmask[i] = nm;
    }
}

// score <- tanh(score) * newmask   (per-node gate)
__global__ void gate_scale(float* __restrict__ score, const float* __restrict__ nm, int n) {
    int i = blockIdx.x * blockDim.x + threadIdx.x;
    if (i < n) score[i] = tanhf(score[i]) * nm[i];
}

__global__ void gate_apply(const float* __restrict__ h, const float* __restrict__ gs,
                           float* __restrict__ out, int n) {
    int id = blockIdx.x * blockDim.x + threadIdx.x;
    if (id >= n) return;
    out[id] = h[id] * gs[id >> 7];
}

// z[g, 0:128] += gmp, z[g, 128:256] += gap; one block (128 thr) per graph
__global__ __launch_bounds__(128)
void readout_kernel(const float* __restrict__ h, const float* __restrict__ nm,
                    const int* __restrict__ starts, const int* __restrict__ counts,
                    float* __restrict__ z) {
    int g = blockIdx.x, f = threadIdx.x;
    int s = starts[g], cnt = counts[g];
    float mx = -INFINITY, sm = 0.0f; int c = 0;
    for (int i = s; i < s + cnt; ++i) {
        if (nm[i] > 0.5f) {
            float v = h[(size_t)i * FD + f];
            mx = fmaxf(mx, v); sm += v; ++c;
        }
    }
    z[(size_t)g * 256 + f]        += (c > 0) ? mx : 0.0f;
    z[(size_t)g * 256 + 128 + f]  += sm / fmaxf((float)c, 1.0f);
}

// logits = t2(500x64) @ Wl3(64x10) + bl3
__global__ void logits_kernel(const float* __restrict__ A, const float* __restrict__ W,
                              const float* __restrict__ b, float* __restrict__ out) {
    int id = blockIdx.x * blockDim.x + threadIdx.x;
    if (id >= NG * NC) return;
    int g = id / NC, c = id - g * NC;
    float acc = b[c];
    for (int k = 0; k < 64; ++k) acc += A[g * 64 + k] * W[k * NC + c];
    out[id] = acc;
}

__global__ void logsoftmax_kernel(const float* __restrict__ in, float* __restrict__ out) {
    int g = blockIdx.x * blockDim.x + threadIdx.x;
    if (g >= NG) return;
    float m = -INFINITY;
    for (int c = 0; c < NC; ++c) m = fmaxf(m, in[g * NC + c]);
    float se = 0.0f;
    for (int c = 0; c < NC; ++c) se += expf(in[g * NC + c] - m);
    float ls = logf(se);
    for (int c = 0; c < NC; ++c) out[g * NC + c] = in[g * NC + c] - m - ls;
}

// ---------------------------------------------------------------------------
extern "C" void kernel_launch(void* const* d_in, const int* in_sizes, int n_in,
                              void* d_out, int out_size, void* d_ws, size_t ws_size,
                              hipStream_t stream) {
    (void)in_sizes; (void)n_in; (void)out_size; (void)ws_size;

    const float* x    = (const float*)d_in[0];
    const float* W[3]  = {(const float*)d_in[1], (const float*)d_in[4], (const float*)d_in[7]};
    const float* bb[3] = {(const float*)d_in[2], (const float*)d_in[5], (const float*)d_in[8]};
    const float* pp[3] = {(const float*)d_in[3], (const float*)d_in[6], (const float*)d_in[9]};
    const float* Wl1 = (const float*)d_in[10]; const float* bl1 = (const float*)d_in[11];
    const float* Wl2 = (const float*)d_in[12]; const float* bl2 = (const float*)d_in[13];
    const float* Wl3 = (const float*)d_in[14]; const float* bl3 = (const float*)d_in[15];
    const int* ei    = (const int*)d_in[16];
    const int* batch = (const int*)d_in[17];
    const int* srcp = ei;
    const int* dstp = ei + NE;

    // workspace carve-out (256B aligned bumps)
    char* base = (char*)d_ws;
    auto take = [&](size_t bytes) {
        char* p = base;
        base += (bytes + 255) & ~(size_t)255;
        return p;
    };
    float*  hW    = (float*)take((size_t)NN * FD * 4);   // gemm output
    float*  agg   = (float*)take((size_t)NN * FD * 4);   // aggregate -> activated h
    float*  hcur  = (float*)take((size_t)NN * FD * 4);   // gated output / next input
    float*  dinv  = (float*)take((size_t)NN * 4);        // deg -> dinv
    float*  maskA = (float*)take((size_t)NN * 4);
    float*  maskB = (float*)take((size_t)NN * 4);
    float*  score = (float*)take((size_t)NN * 4);
    float*  pn    = (float*)take(256);
    int*    starts= (int*)take((size_t)NG * 4);
    int*    counts= (int*)take((size_t)NG * 4);
    float*  z     = (float*)take((size_t)NG * 256 * 4);
    float*  t1    = (float*)take((size_t)NG * 128 * 4);
    float*  t2    = (float*)take((size_t)NG * 64 * 4);
    float*  lg    = (float*)take((size_t)NG * NC * 4);
    bf16_t* Wbf   = (bf16_t*)take((size_t)256 * 128 * 2);  // packed B (max 64KB)

    hipMemsetAsync(z, 0, (size_t)NG * 256 * 4, stream);
    fill_f32<<<(NN + 255) / 256, 256, 0, stream>>>(maskA, 1.0f, NN);
    graph_ranges<<<(NG + 255) / 256, 256, 0, stream>>>(batch, NN, starts, counts);

    const float* Xin = x;
    float* mcur = maskA; float* mnew = maskB;

    for (int l = 0; l < 3; ++l) {
        // h = Xin @ W[l]  (bf16 WMMA, f32 accumulate)
        pack_b_bf16<<<(FD * FD + 255) / 256, 256, 0, stream>>>(W[l], Wbf, FD, FD, FD / 32);
        wmma_gemm<FD, false, false><<<dim3(NN / 16, 1), 256, 0, stream>>>(
            Xin, (const v16bf*)Wbf, hW, NN, FD, nullptr);

        // degree -> dinv
        hipMemsetAsync(dinv, 0, (size_t)NN * 4, stream);
        edge_deg<<<(NE + 255) / 256, 256, 0, stream>>>(srcp, dstp, mcur, dinv, NE);
        node_dinv<<<(NN + 255) / 256, 256, 0, stream>>>(dinv, mcur, NN);

        // neighbor aggregation (dominant cost: ~1.6 GB of loads+atomics)
        hipMemsetAsync(agg, 0, (size_t)NN * FD * 4, stream);
        edge_agg<<<(NE + 7) / 8, 256, 0, stream>>>(srcp, dstp, dinv, mcur, hW, agg, NE);

        // agg <- relu(agg + self-loop + b)
        gcn_finalize<<<(NN * FD + 255) / 256, 256, 0, stream>>>(
            agg, hW, dinv, mcur, bb[l], NN * FD);

        // top-k pooling
        pnorm_kernel<<<1, 128, 0, stream>>>(pp[l], pn);
        score_kernel<<<(NN + 7) / 8, 256, 0, stream>>>(agg, pp[l], pn, score, NN);
        pool_kernel<<<NG, 256, 0, stream>>>(score, mcur, starts, counts, mnew);
        gate_scale<<<(NN + 255) / 256, 256, 0, stream>>>(score, mnew, NN);
        gate_apply<<<(NN * FD + 255) / 256, 256, 0, stream>>>(agg, score, hcur, NN * FD);

        // readout accumulates into z
        readout_kernel<<<NG, 128, 0, stream>>>(hcur, mnew, starts, counts, z);

        float* tmp = mcur; mcur = mnew; mnew = tmp;
        Xin = hcur;
    }

    // MLP head: two WMMA layers + tiny logits/log_softmax
    pack_b_bf16<<<(256 * 128 + 255) / 256, 256, 0, stream>>>(Wl1, Wbf, 256, 128, 256 / 32);
    wmma_gemm<256, true, true><<<dim3((NG + 15) / 16, 1), 256, 0, stream>>>(
        z, (const v16bf*)Wbf, t1, NG, 128, bl1);
    pack_b_bf16<<<(128 * 64 + 255) / 256, 256, 0, stream>>>(Wl2, Wbf, 128, 64, 128 / 32);
    wmma_gemm<128, true, true><<<dim3((NG + 15) / 16, 1), 256, 0, stream>>>(
        t1, (const v16bf*)Wbf, t2, NG, 64, bl2);
    logits_kernel<<<(NG * NC + 255) / 256, 256, 0, stream>>>(t2, Wl3, bl3, lg);
    logsoftmax_kernel<<<(NG + 255) / 256, 256, 0, stream>>>(lg, (float*)d_out);
}
